// RelPartialLearnableMultiHeadAttn_72619307041089
// MI455X (gfx1250) — matrix operand order repaired and checked by reading, not compiled, not run on hardware
//
#include <hip/hip_runtime.h>
#include <hip/hip_bf16.h>

#define QLEN 1024
#define BSZ  4
#define EDIM 1024
#define NH   16
#define DH   64
#define MROWS (QLEN * BSZ)   /* 4096 rows (i-major, then b) */
#define NQKV  (3 * NH * DH)  /* 3072  */

typedef __attribute__((ext_vector_type(16))) _Float16 v16h;
typedef __attribute__((ext_vector_type(8)))  _Float16 v8h;
typedef __attribute__((ext_vector_type(8)))  float    v8f;
typedef __attribute__((ext_vector_type(4)))  unsigned int u32x4;
typedef __attribute__((ext_vector_type(4)))  int      i32x4;
typedef __attribute__((ext_vector_type(8)))  int      i32x8;

#define V8FZERO ((v8f){0.f,0.f,0.f,0.f,0.f,0.f,0.f,0.f})

// Tensor Data Mover availability (probe-confirmed on both toolchains, arity differs)
#if defined(__has_builtin)
#  if __has_builtin(__builtin_amdgcn_tensor_load_to_lds) && \
      __has_builtin(__builtin_amdgcn_s_wait_tensorcnt)
#    define USE_TDM 1
#  endif
#endif
#ifndef USE_TDM
#  define USE_TDM 0
#endif

// ---------------------------------------------------------------- WMMA core
static __device__ inline v8f wmma16(v16h a, v16h b, v8f c) {
  // D = A(16x32 f16) * B(32x16 f16) + C(16x16 f32)
  return __builtin_amdgcn_wmma_f32_16x16x32_f16(false, a, false, b,
                                                (short)0, c, false, false);
}

// A fragment 16(M)x32(K), K contiguous, row stride ldk (halves).
// lanes0-15 row=l, K 0..15; lanes16-31 row=l-16, K 16..31 (contiguous)
static __device__ inline v16h load_a16x32(const _Float16* __restrict__ base,
                                          int ldk, int lane) {
  const int row = lane & 15;
  const int kb  = (lane < 16) ? 0 : 16;
  const _Float16* p = base + row * ldk + kb;
  v8h lo = *(const v8h*)(p);
  v8h hi = *(const v8h*)(p + 8);
  return __builtin_shufflevector(lo, hi, 0,1,2,3,4,5,6,7,8,9,10,11,12,13,14,15);
}

// B fragment 32(K)x16(N) from a [N][K] (transposed) source:
// lanes0-15 col=l hold K 0..15 contiguously; lanes16-31 col=l-16 hold K 16..31.
static __device__ inline v16h load_bT32x16(const _Float16* __restrict__ base,
                                           int ldk, int lane) {
  const int col = lane & 15;
  const int kb  = (lane < 16) ? 0 : 16;
  return *(const v16h*)(base + col * ldk + kb);
}

// wave32 row reductions: xor masks 1,2,4,8 stay within each 16-lane half,
// matching one row-group of the f32 C fragment.
static __device__ inline float rmax16(float v) {
  v = fmaxf(v, __shfl_xor(v, 1, 32));
  v = fmaxf(v, __shfl_xor(v, 2, 32));
  v = fmaxf(v, __shfl_xor(v, 4, 32));
  v = fmaxf(v, __shfl_xor(v, 8, 32));
  return v;
}
static __device__ inline float rsum16(float v) {
  v += __shfl_xor(v, 1, 32);
  v += __shfl_xor(v, 2, 32);
  v += __shfl_xor(v, 4, 32);
  v += __shfl_xor(v, 8, 32);
  return v;
}

#if USE_TDM
// Issue one TDM 2-D tile load: tile_k x tile_rows f16 elements, row stride
// tensor_w elements, from gaddr into wave-local LDS at lds_byte_off.
// D# bitfields per CDNA5 ISA ch.8 (group0: count/lds/global/type, group1:
// data_size + dims/strides). Groups 2/3 zero (<=2D tensor).
static __device__ inline void tdm_load_tile(unsigned lds_byte_off,
                                            const void* gaddr,
                                            int tile_k, int tile_rows,
                                            int tensor_w, int tensor_h) {
  unsigned long long ga = (unsigned long long)(size_t)gaddr;
  u32x4 g0;
  g0[0] = 1u;                                             // count=1, user mode
  g0[1] = lds_byte_off;                                   // lds_addr (bytes)
  g0[2] = (unsigned)(ga & 0xffffffffu);                   // global_addr[31:0]
  g0[3] = (unsigned)((ga >> 32) & 0x01ffffffu) | (2u << 30); // addr[56:32],type=2
  i32x8 g1;
  g1[0] = (1 << 16);                                      // data_size=1 -> 2B
  g1[1] = (tensor_w & 0xffff) << 16;                      // tensor_dim0[15:0]
  g1[2] = ((unsigned)tensor_w >> 16) | ((tensor_h & 0xffff) << 16);
  g1[3] = ((unsigned)tensor_h >> 16) | (tile_k << 16);    // tile_dim0
  g1[4] = tile_rows;                                      // tile_dim1 (dim2=0)
  g1[5] = tensor_w;                                       // dim0_stride[31:0]
  g1[6] = 0;
  g1[7] = 0;
  i32x4 z4 = (i32x4){0, 0, 0, 0};
#if __clang_major__ >= 23
  i32x8 z8 = (i32x8){0, 0, 0, 0, 0, 0, 0, 0};
  __builtin_amdgcn_tensor_load_to_lds(g0, g1, z4, z4, z8, 0);
#else
  __builtin_amdgcn_tensor_load_to_lds(g0, g1, z4, z4, 0);
#endif
}
#endif

// Shared GEMM mainloop: 16 rows (one wave) x 64 cols, K = EDIM.
// TDM path: A tile (16x64 f16) staged per-wave into double-buffered LDS slab
// by the Tensor Data Mover while WMMA consumes the other buffer; B fragments
// stream straight from global (contiguous 32B chunks feed v_wmma directly).
static __device__ inline void gemm_mainloop(const _Float16* __restrict__ Arow,
                                            const _Float16* __restrict__ Wbase,
                                            _Float16* aslab, int lane,
                                            v8f acc[4]) {
#if USE_TDM
  const unsigned lds0 = (unsigned)(size_t)(const void*)aslab;  // LDS offset
  tdm_load_tile(lds0, Arow, 64, 16, EDIM, 16);
  for (int kc = 0; kc < EDIM / 64; ++kc) {
    if (kc + 1 < EDIM / 64) {
      tdm_load_tile(lds0 + (unsigned)(((kc + 1) & 1) * 2048),
                    Arow + (kc + 1) * 64, 64, 16, EDIM, 16);
      __builtin_amdgcn_s_wait_tensorcnt(1);   // previous tile landed (in-order)
    } else {
      __builtin_amdgcn_s_wait_tensorcnt(0);   // drain
    }
    const _Float16* lbase = aslab + (kc & 1) * 1024;
#pragma unroll
    for (int ks = 0; ks < 2; ++ks) {
      v16h a = load_a16x32(lbase + ks * 32, 64, lane);
#pragma unroll
      for (int t = 0; t < 4; ++t) {
        v16h b = load_bT32x16(Wbase + t * 16 * EDIM + kc * 64 + ks * 32,
                              EDIM, lane);
        acc[t] = wmma16(a, b, acc[t]);
      }
    }
  }
#else
  (void)aslab;
  for (int k0 = 0; k0 < EDIM; k0 += 32) {
    __builtin_prefetch(Arow + k0 + 128, 0, 3);
    v16h a = load_a16x32(Arow + k0, EDIM, lane);
#pragma unroll
    for (int t = 0; t < 4; ++t) {
      v16h b = load_bT32x16(Wbase + t * 16 * EDIM + k0, EDIM, lane);
      acc[t] = wmma16(a, b, acc[t]);
    }
  }
#endif
}

// ---------------------------------------------------------------- convert
__global__ __launch_bounds__(256)
void cvt_f32_f16(const float* __restrict__ src, _Float16* __restrict__ dst, int n) {
  int i = blockIdx.x * blockDim.x + threadIdx.x;
  if (i < n) dst[i] = (_Float16)src[i];
}

// ---------------------------------------------------------------- QKV GEMM
// C[m][nf] = sum_e act[m][e] * W[nf][e];  m = i*BSZ + b, nf = which*1024 + h*64 + dh
__global__ __launch_bounds__(128)
void qkv_gemm(const _Float16* __restrict__ A, const _Float16* __restrict__ W,
              const float* __restrict__ rwb, const float* __restrict__ rrb,
              _Float16* __restrict__ qrw, _Float16* __restrict__ qrr,
              _Float16* __restrict__ kout, _Float16* __restrict__ vt) {
  __shared__ alignas(32) _Float16 Asl[4][2][1024];   // per-wave TDM slabs
  const int lane = threadIdx.x & 31;
  const int wid  = threadIdx.x >> 5;
  const int m0 = blockIdx.x * 64 + wid * 16;
  const int n0 = blockIdx.y * 64;
  v8f acc[4];
#pragma unroll
  for (int t = 0; t < 4; ++t) acc[t] = V8FZERO;

  gemm_mainloop(A + (size_t)m0 * EDIM, W + (size_t)n0 * EDIM,
                &Asl[wid][0][0], lane, acc);

  const int jl = lane & 15;
  const int rhalf = (lane >> 4) << 3;
#pragma unroll
  for (int t = 0; t < 4; ++t) {
    const int nf  = n0 + t * 16 + jl;
    const int which = nf >> 10;      // 0=q 1=k 2=v
    const int rem = nf & 1023;
    const int hd  = rem >> 6;
    const int dh  = rem & 63;
#pragma unroll
    for (int c = 0; c < 8; ++c) {
      const int m = m0 + c + rhalf;
      const int i = m >> 2;          // BSZ == 4
      const int b = m & 3;
      const int hb = b * NH + hd;
      const float v = acc[t][c];
      if (which == 0) {
        qrw[(hb * QLEN + i) * DH + dh] = (_Float16)(v + rwb[hd * DH + dh]);
        qrr[(hb * QLEN + i) * DH + dh] = (_Float16)(v + rrb[hd * DH + dh]);
      } else if (which == 1) {
        kout[(hb * QLEN + i) * DH + dh] = (_Float16)v;
      } else {
        vt[(hb * DH + dh) * QLEN + i] = (_Float16)v;  // V stored transposed
      }
    }
  }
}

// ---------------------------------------------------------------- r GEMM
__global__ __launch_bounds__(128)
void r_gemm(const _Float16* __restrict__ A, const _Float16* __restrict__ W,
            _Float16* __restrict__ rk) {
  __shared__ alignas(32) _Float16 Asl[4][2][1024];
  const int lane = threadIdx.x & 31;
  const int wid  = threadIdx.x >> 5;
  const int m0 = blockIdx.x * 64 + wid * 16;
  const int n0 = blockIdx.y * 64;
  v8f acc[4];
#pragma unroll
  for (int t = 0; t < 4; ++t) acc[t] = V8FZERO;

  gemm_mainloop(A + (size_t)m0 * EDIM, W + (size_t)n0 * EDIM,
                &Asl[wid][0][0], lane, acc);

  const int jl = lane & 15;
  const int rhalf = (lane >> 4) << 3;
#pragma unroll
  for (int t = 0; t < 4; ++t) {
    const int nf = n0 + t * 16 + jl;
    const int hd = nf >> 6, dh = nf & 63;
#pragma unroll
    for (int c = 0; c < 8; ++c) {
      const int m = m0 + c + rhalf;
      rk[(hd * QLEN + m) * DH + dh] = (_Float16)acc[t][c];
    }
  }
}

// ---------------------------------------------------------------- fused attention
// grid (QLEN/64, NH, BSZ), block 128 (4 independent waves; 16 query rows each)
__global__ __launch_bounds__(128)
void rel_attn(const _Float16* __restrict__ qrw_r, const _Float16* __restrict__ qrr_r,
              const _Float16* __restrict__ k_r,  const _Float16* __restrict__ vt_r,
              const _Float16* __restrict__ qrw_p, const _Float16* __restrict__ qrr_p,
              const _Float16* __restrict__ k_p,  const _Float16* __restrict__ vt_p,
              const _Float16* __restrict__ rk,
              _Float16* __restrict__ vec_r, _Float16* __restrict__ vec_p) {
  __shared__ float bdr_s[4][3][256];
  __shared__ float bdp_s[4][3][256];
  __shared__ alignas(32) _Float16 p_s[4][512];

  const int lane = threadIdx.x & 31;
  const int wid  = threadIdx.x >> 5;
  const int hd = blockIdx.y, b = blockIdx.z;
  const int i0 = blockIdx.x * 64 + wid * 16;
  const int hb = b * NH + hd;
  const _Float16* Qwr = qrw_r + (size_t)hb * QLEN * DH;
  const _Float16* Qrr = qrr_r + (size_t)hb * QLEN * DH;
  const _Float16* Qwp = qrw_p + (size_t)hb * QLEN * DH;
  const _Float16* Qrp = qrr_p + (size_t)hb * QLEN * DH;
  const _Float16* Kr  = k_r   + (size_t)hb * QLEN * DH;
  const _Float16* Kp  = k_p   + (size_t)hb * QLEN * DH;
  const _Float16* Vr  = vt_r  + (size_t)hb * DH * QLEN;  // [dh][q]
  const _Float16* Vp  = vt_p  + (size_t)hb * DH * QLEN;
  const _Float16* Rk  = rk    + (size_t)hd * QLEN * DH;

  v16h aWr[2], aWp[2], aRr[2], aRp[2];
#pragma unroll
  for (int ks = 0; ks < 2; ++ks) {
    aWr[ks] = load_a16x32(Qwr + i0 * DH + ks * 32, DH, lane);
    aWp[ks] = load_a16x32(Qwp + i0 * DH + ks * 32, DH, lane);
    aRr[ks] = load_a16x32(Qrr + i0 * DH + ks * 32, DH, lane);
    aRp[ks] = load_a16x32(Qrp + i0 * DH + ks * 32, DH, lane);
  }
  v8f o_r[4], o_p[4];
  float mrun[8], lrun[8];
#pragma unroll
  for (int t = 0; t < 4; ++t) { o_r[t] = V8FZERO; o_p[t] = V8FZERO; }
#pragma unroll
  for (int c = 0; c < 8; ++c) { mrun[c] = -3.0e38f; lrun[c] = 0.f; }

  const int jl = lane & 15;
  const int rhalf = (lane >> 4) << 3;  // 0 or 8
  const int kb16 = (lane < 16) ? 0 : 16;

  for (int j0 = 0; j0 <= i0 + 15; j0 += 32) {
    // ---- AC content scores: (q+rwb)·K via WMMA
    v8f acr[2], acp[2];
#pragma unroll
    for (int s = 0; s < 2; ++s) {
      const int jb = j0 + 16 * s;
      v8f t1 = V8FZERO, t2 = V8FZERO, t3 = V8FZERO;
#pragma unroll
      for (int ks = 0; ks < 2; ++ks) {
        v16h kr = load_bT32x16(Kr + jb * DH + ks * 32, DH, lane);
        v16h kp = load_bT32x16(Kp + jb * DH + ks * 32, DH, lane);
        t1 = wmma16(aWr[ks], kr, t1);   // rwq_r . k_r
        t2 = wmma16(aWp[ks], kp, t2);   // rwq_p . k_p
        t3 = wmma16(aWr[ks], kp, t3);   // rwq_r . k_p
      }
      acr[s] = t1 - t2;
      acp[s] = t3 + t3;                 // faithful 2x term
    }

    // ---- BD in (i, d=i-j) space. rel_shift => BD[i,j] = (q+rrb)[i]·rk[1023-(i-j)]
    // d-window of a 16x32 tile is exactly 3 aligned 16-tiles.
    const int dbase0 = i0 - j0 - 32;
#pragma unroll
    for (int t = 0; t < 3; ++t) {
      v8f br = V8FZERO, bp = V8FZERO;
      int d = dbase0 + t * 16 + jl;
      int srow = 1023 - d;
      srow = srow < 0 ? 0 : (srow > 1023 ? 1023 : srow);
#pragma unroll
      for (int ks = 0; ks < 2; ++ks) {
        v16h rb = *(const v16h*)(Rk + srow * DH + ks * 32 + kb16);
        br = wmma16(aRr[ks], rb, br);
        bp = wmma16(aRp[ks], rb, bp);
      }
#pragma unroll
      for (int c = 0; c < 8; ++c) {
        const int rl = c + rhalf;
        bdr_s[wid][t][rl * 16 + jl] = br[c];
        bdp_s[wid][t][rl * 16 + jl] = bp[c];
      }
    }
    // same-wave DS ops are in-order; waves are independent (no barrier)

    // ---- score + online softmax
    float rowscale[8];
#pragma unroll
    for (int c = 0; c < 8; ++c) {
      const int rl = c + rhalf;
      const int i = i0 + rl;
      float sc[2];
#pragma unroll
      for (int s = 0; s < 2; ++s) {
        const int j  = j0 + 16 * s + jl;
        const int dd = (i - j) - dbase0;     // in [1,47] for this tile
        const int tt = dd >> 4;
        const int dl = dd & 15;
        const float br = bdr_s[wid][tt][rl * 16 + dl];
        const float bp = bdp_s[wid][tt][rl * 16 + dl];
        const float ar = acr[s][c];
        const float ap = acp[s][c];
        float v = (sqrtf(ar * ar + ap * ap) + sqrtf(br * br + bp * bp)) * 0.125f;
        sc[s] = (j > i) ? -1.0e30f : v;      // causal mask
      }
      const float rm  = rmax16(fmaxf(sc[0], sc[1]));
      const float mn  = fmaxf(mrun[c], rm);
      const float scl = __expf(mrun[c] - mn);
      const float p0  = __expf(sc[0] - mn);
      const float p1  = __expf(sc[1] - mn);
      lrun[c] = lrun[c] * scl + rsum16(p0 + p1);
      mrun[c] = mn;
      rowscale[c] = scl;
      p_s[wid][rl * 32 + jl]      = (_Float16)p0;
      p_s[wid][rl * 32 + 16 + jl] = (_Float16)p1;
    }
#pragma unroll
    for (int t = 0; t < 4; ++t)
#pragma unroll
      for (int c = 0; c < 8; ++c) { o_r[t][c] *= rowscale[c]; o_p[t][c] *= rowscale[c]; }

    // ---- PV: P re-laid from C-layout to A-fragment via per-wave LDS
    v16h pa = load_a16x32(&p_s[wid][0], 32, lane);
#pragma unroll
    for (int t = 0; t < 4; ++t) {
      const int ncol = t * 16 + jl;
      v16h vr = *(const v16h*)(Vr + ncol * QLEN + j0 + kb16);
      v16h vp = *(const v16h*)(Vp + ncol * QLEN + j0 + kb16);
      o_r[t] = wmma16(pa, vr, o_r[t]);
      o_p[t] = wmma16(pa, vp, o_p[t]);
    }
  }

  // ---- epilogue: normalize, scatter to [m][NH*DH] f16 for output GEMM
#pragma unroll
  for (int c = 0; c < 8; ++c) {
    const float inv = 1.0f / lrun[c];
    const int rl = c + rhalf;
    const int i = i0 + rl;
    const int mrow = i * BSZ + b;
#pragma unroll
    for (int t = 0; t < 4; ++t) {
      const int col = hd * DH + t * 16 + jl;
      vec_r[mrow * (NH * DH) + col] = (_Float16)(o_r[t][c] * inv);
      vec_p[mrow * (NH * DH) + col] = (_Float16)(o_p[t][c] * inv);
    }
  }
}

// ---------------------------------------------------------------- output GEMM + residual
__global__ __launch_bounds__(128)
void out_gemm(const _Float16* __restrict__ A, const _Float16* __restrict__ W,
              const float* __restrict__ resid, float* __restrict__ out) {
  __shared__ alignas(32) _Float16 Asl[4][2][1024];
  const int lane = threadIdx.x & 31;
  const int wid  = threadIdx.x >> 5;
  const int m0 = blockIdx.x * 64 + wid * 16;
  const int n0 = blockIdx.y * 64;
  v8f acc[4];
#pragma unroll
  for (int t = 0; t < 4; ++t) acc[t] = V8FZERO;

  gemm_mainloop(A + (size_t)m0 * EDIM, W + (size_t)n0 * EDIM,
                &Asl[wid][0][0], lane, acc);

  const int jl = lane & 15;
  const int rhalf = (lane >> 4) << 3;
#pragma unroll
  for (int t = 0; t < 4; ++t) {
    const int col = n0 + t * 16 + jl;
#pragma unroll
    for (int c = 0; c < 8; ++c) {
      const int m = m0 + c + rhalf;
      out[(size_t)m * EDIM + col] = acc[t][c] + resid[(size_t)m * EDIM + col];
    }
  }
}

// ---------------------------------------------------------------- in-place LayerNorm
__global__ __launch_bounds__(256)
void ln_inplace(float* __restrict__ out, const float* __restrict__ g,
                const float* __restrict__ bb) {
  const int row = blockIdx.x;
  float* x = out + (size_t)row * EDIM;
  __shared__ float ssum[256], ssq[256];
  float s = 0.f, q = 0.f;
  for (int k = threadIdx.x; k < EDIM; k += 256) { float v = x[k]; s += v; q += v * v; }
  ssum[threadIdx.x] = s; ssq[threadIdx.x] = q;
  __syncthreads();
  for (int off = 128; off > 0; off >>= 1) {
    if (threadIdx.x < off) {
      ssum[threadIdx.x] += ssum[threadIdx.x + off];
      ssq[threadIdx.x]  += ssq[threadIdx.x + off];
    }
    __syncthreads();
  }
  const float mu  = ssum[0] * (1.0f / EDIM);
  const float var = ssq[0] * (1.0f / EDIM) - mu * mu;
  const float rs  = rsqrtf(var + 1e-5f);
  for (int k = threadIdx.x; k < EDIM; k += 256)
    x[k] = (x[k] - mu) * rs * g[k] + bb[k];
}

// ---------------------------------------------------------------- host
extern "C" void kernel_launch(void* const* d_in, const int* in_sizes, int n_in,
                              void* d_out, int out_size, void* d_ws, size_t ws_size,
                              hipStream_t stream) {
  const float* w_real  = (const float*)d_in[0];
  const float* w_phase = (const float*)d_in[1];
  const float* r_in    = (const float*)d_in[2];
  const float* rwb     = (const float*)d_in[3];
  const float* rrb     = (const float*)d_in[4];
  const float* W_qkv   = (const float*)d_in[5];
  const float* W_qkv1  = (const float*)d_in[6];
  const float* W_r     = (const float*)d_in[7];
  const float* W_o     = (const float*)d_in[8];
  const float* W_o1    = (const float*)d_in[9];
  const float* gamma   = (const float*)d_in[10];
  const float* beta    = (const float*)d_in[11];
  // d_in[12] = attn_mask (causal; handled analytically)
  float* out = (float*)d_out;

  char* ws = (char*)d_ws;
  size_t off = 0;
  auto alloc = [&](size_t bytes) -> void* {
    void* p = ws + off;
    off = (off + bytes + 255) & ~(size_t)255;
    return p;
  };
  const size_t HBYTES_ACT  = (size_t)MROWS * EDIM * 2;          // 8 MB
  const size_t HBYTES_HEAD = (size_t)BSZ * NH * QLEN * DH * 2;  // 8 MB
  _Float16* Ar    = (_Float16*)alloc(HBYTES_ACT);
  _Float16* Ap    = (_Float16*)alloc(HBYTES_ACT);
  _Float16* Wq_h  = (_Float16*)alloc((size_t)NQKV * EDIM * 2);
  _Float16* Wq1_h = (_Float16*)alloc((size_t)NQKV * EDIM * 2);
  _Float16* Wr_h  = (_Float16*)alloc((size_t)EDIM * EDIM * 2);
  _Float16* R_h   = (_Float16*)alloc((size_t)QLEN * EDIM * 2);
  _Float16* Wo_h  = (_Float16*)alloc((size_t)EDIM * EDIM * 2);
  _Float16* Wo1_h = (_Float16*)alloc((size_t)EDIM * EDIM * 2);
  _Float16* Qwr = (_Float16*)alloc(HBYTES_HEAD);
  _Float16* Qrr = (_Float16*)alloc(HBYTES_HEAD);
  _Float16* Krh = (_Float16*)alloc(HBYTES_HEAD);
  _Float16* Vtr = (_Float16*)alloc(HBYTES_HEAD);
  _Float16* Qwp = (_Float16*)alloc(HBYTES_HEAD);
  _Float16* Qrp = (_Float16*)alloc(HBYTES_HEAD);
  _Float16* Kph = (_Float16*)alloc(HBYTES_HEAD);
  _Float16* Vtp = (_Float16*)alloc(HBYTES_HEAD);
  _Float16* Rk_h = (_Float16*)alloc((size_t)NH * QLEN * DH * 2);
  _Float16* VecR = (_Float16*)alloc(HBYTES_ACT);
  _Float16* VecP = (_Float16*)alloc(HBYTES_ACT);

  auto cvt = [&](const float* s, _Float16* d, int n) {
    cvt_f32_f16<<<(n + 255) / 256, 256, 0, stream>>>(s, d, n);
  };
  cvt(w_real,  Ar,    MROWS * EDIM);
  cvt(w_phase, Ap,    MROWS * EDIM);
  cvt(W_qkv,   Wq_h,  NQKV * EDIM);
  cvt(W_qkv1,  Wq1_h, NQKV * EDIM);
  cvt(W_r,     Wr_h,  EDIM * EDIM);
  cvt(r_in,    R_h,   QLEN * EDIM);
  cvt(W_o,     Wo_h,  EDIM * EDIM);
  cvt(W_o1,    Wo1_h, EDIM * EDIM);

  qkv_gemm<<<dim3(MROWS / 64, NQKV / 64), 128, 0, stream>>>(
      Ar, Wq_h, rwb, rrb, Qwr, Qrr, Krh, Vtr);
  qkv_gemm<<<dim3(MROWS / 64, NQKV / 64), 128, 0, stream>>>(
      Ap, Wq1_h, rwb, rrb, Qwp, Qrp, Kph, Vtp);
  r_gemm<<<dim3(QLEN / 64, EDIM / 64), 128, 0, stream>>>(R_h, Wr_h, Rk_h);

  rel_attn<<<dim3(QLEN / 64, NH, BSZ), 128, 0, stream>>>(
      Qwr, Qrr, Krh, Vtr, Qwp, Qrp, Kph, Vtp, Rk_h, VecR, VecP);

  out_gemm<<<dim3(MROWS / 64, EDIM / 64), 128, 0, stream>>>(
      VecR, Wo_h, w_real, out);
  out_gemm<<<dim3(MROWS / 64, EDIM / 64), 128, 0, stream>>>(
      VecP, Wo1_h, w_phase, out + (size_t)MROWS * EDIM);

  ln_inplace<<<dim3(2 * MROWS), 256, 0, stream>>>(out, gamma, beta);
}
